// SpatialTransformer_60782377173633
// MI455X (gfx1250) — matrix operand order repaired and checked by reading, not compiled, not run on hardware
//
#include <hip/hip_runtime.h>
#include <hip/hip_bf16.h>

typedef float    v2f  __attribute__((ext_vector_type(2)));
typedef float    v8f  __attribute__((ext_vector_type(8)));
typedef _Float16 v16h __attribute__((ext_vector_type(16)));

// Problem geometry (fixed by the reference)
#define BATCH 16
#define IH    224
#define IW    224
#define IC    16
#define OC    8            // conv output channels (padded to 16 in WMMA N)
#define CW    218          // conv out W (VALID, k=7)
#define PH    109          // pooled H
#define PW    109          // pooled W
#define FLAT  (PH*PW*OC)   // 95048
#define NPOS  49           // 7*7 filter positions
#define NCHUNK 25          // ceil(784/32) K-chunks of 32 (K padded 784 -> 800)
#define WFRAG_DW (NCHUNK*256)   // 6400 dwords of packed f16 B-fragments

// ---------------------------------------------------------------------------
// Kernel 0: zero the FC1 accumulator (atomics target) each launch.
// ---------------------------------------------------------------------------
__global__ void stn_zero(float* __restrict__ acc) {
    int t = threadIdx.x;
    if (t < 512) acc[t] = 0.0f;
}

// ---------------------------------------------------------------------------
// Kernel W: pre-format conv weights into f16 WMMA B-fragments.
// B (32x16, f16) per-lane layout: VGPR j<4, lanes 0-15 -> K=2j,2j+1 ([15:0],[31:16]),
// j>=4 -> K=16+2(j-4),+1; lanes 16-31 add 8.  N = lane&15 (oc >= 8 zero-padded).
// Chunk c covers global K = 32c..32c+31 = positions 2c,2c+1 (16 channels each);
// position 49 (chunk 24 upper half) is zero padding.
// wfrag[c*256 + lane*8 + j] = packed pair.
// ---------------------------------------------------------------------------
__global__ void stn_wfmt(const float* __restrict__ conv_w,   // [7,7,16,8]
                         unsigned* __restrict__ wfrag)       // [6400]
{
    int idx = blockIdx.x * 256 + threadIdx.x;   // 25*256 = 6400 exact
    int c  = idx >> 8;
    int r  = idx & 255;
    int l  = r >> 3;                 // lane
    int j  = r & 7;                  // fragment vgpr
    int kk = ((j >> 2) << 4) + ((l >> 4) << 3) + 2 * (j & 3);
    int oc = l & 15;
    int K0 = 32 * c + kk;
    int pos = K0 >> 4;               // ky*7+kx
    int ch  = K0 & 15;               // even, ch+1 in same position
    float v0 = 0.0f, v1 = 0.0f;
    if (oc < OC && pos < NPOS) {
        v0 = conv_w[(pos * IC + ch    ) * OC + oc];
        v1 = conv_w[(pos * IC + ch + 1) * OC + oc];
    }
    union { _Float16 h[2]; unsigned u; } pk;
    pk.h[0] = (_Float16)v0;
    pk.h[1] = (_Float16)v1;
    wfrag[idx] = pk.u;
}

// ---------------------------------------------------------------------------
// Kernel 1: Conv7x7 + bias + ReLU + MaxPool2x2, implicit GEMM with
// V_WMMA_F32_16X16X32_F16 (f32 accumulate).  One wave = 2 conv rows x 8 cols.
// B-fragments are DMA'd into LDS via global_load_async_to_lds_b128.
//
// f16 A layout (16x32): lanes 0-15 (M=lane) hold K=0-7 (V0-3), K=16-23 (V4-7);
// lanes 16-31 hold K=8-15, K=24-31.  With chunk c = positions p0=2c, p1=2c+1:
// per lane, element e<8 = x[p0][8*hi + e], e>=8 = x[p1][8*hi + e-8].
// ---------------------------------------------------------------------------
__global__ __launch_bounds__(256) void stn_conv_pool(
    const float* __restrict__ x,        // [16,224,224,16]
    const unsigned* __restrict__ wfrag, // [6400] preformatted f16 fragments
    const float* __restrict__ conv_b,   // [8]
    float* __restrict__ pooled)         // [16,109,109,8]
{
    __shared__ unsigned lds_w[WFRAG_DW];   // 25600 B
    __shared__ float lds_pool[8 * 256];    //  8192 B

    const int tid = threadIdx.x;

    // --- async DMA of formatted weights into LDS (16 B per lane per op) ---
    for (int i = tid; i < WFRAG_DW / 4; i += 256) {
        unsigned lofs = (unsigned)(size_t)(&lds_w[i * 4]);
        unsigned long long ga = (unsigned long long)(size_t)(wfrag + i * 4);
        asm volatile("global_load_async_to_lds_b128 %0, %1, off"
                     :: "v"(lofs), "v"(ga) : "memory");
    }
    asm volatile("s_wait_asynccnt 0x0" ::: "memory");
    __syncthreads();

    const int lane = tid & 31;
    const int wave = tid >> 5;
    const int wg   = blockIdx.x * 8 + wave;   // 0 .. 48831
    const int tx   = wg % 28;                 // column tile (8 conv cols)
    const int tmp  = wg / 28;
    const int py   = tmp % PH;                // pooled row
    const int b    = tmp / PH;                // batch

    const int m   = lane & 15;
    const int row = m >> 3;                   // conv row in tile (0/1)
    const int col = m & 7;                    // conv col in tile
    const int hi  = lane >> 4;
    const int cy  = 2 * py + row;             // conv y (0..217)
    int cx = 8 * tx + col;
    if (cx > CW - 1) cx = CW - 1;             // clamp ragged right edge (discarded)

    const float* xbase = x + (((size_t)b * IH + cy) * IW + cx) * IC + 8 * hi;

    v8f acc = {};
    #pragma unroll 2
    for (int c = 0; c < NCHUNK; ++c) {
        int p0 = 2 * c;
        int p1 = p0 + 1;
        if (p1 > NPOS - 1) p1 = NPOS - 1;     // A garbage x B zero-pad = 0
        int ky0 = p0 / 7, kx0 = p0 - 7 * ky0;
        int ky1 = p1 / 7, kx1 = p1 - 7 * ky1;
        const float* xp0 = xbase + (ky0 * IW + kx0) * IC;
        const float* xp1 = xbase + (ky1 * IW + kx1) * IC;

        float4 f0 = *reinterpret_cast<const float4*>(xp0);
        float4 f1 = *reinterpret_cast<const float4*>(xp0 + 4);
        float4 f2 = *reinterpret_cast<const float4*>(xp1);
        float4 f3 = *reinterpret_cast<const float4*>(xp1 + 4);

        v16h a;
        a[0]  = (_Float16)f0.x; a[1]  = (_Float16)f0.y;
        a[2]  = (_Float16)f0.z; a[3]  = (_Float16)f0.w;
        a[4]  = (_Float16)f1.x; a[5]  = (_Float16)f1.y;
        a[6]  = (_Float16)f1.z; a[7]  = (_Float16)f1.w;
        a[8]  = (_Float16)f2.x; a[9]  = (_Float16)f2.y;
        a[10] = (_Float16)f2.z; a[11] = (_Float16)f2.w;
        a[12] = (_Float16)f3.x; a[13] = (_Float16)f3.y;
        a[14] = (_Float16)f3.z; a[15] = (_Float16)f3.w;

        const uint4* lp = reinterpret_cast<const uint4*>(lds_w + c * 256 + lane * 8);
        union { uint4 q[2]; v16h h; } u;
        u.q[0] = lp[0];
        u.q[1] = lp[1];

        acc = __builtin_amdgcn_wmma_f32_16x16x32_f16(
            false, a, false, u.h, (short)0, acc, false, false);
    }

    // --- spill D tile to wave-private LDS, then 2x2 maxpool + bias + relu ---
    float* pl = lds_pool + wave * 256;
    #pragma unroll
    for (int i = 0; i < 8; ++i) {
        pl[(i + 8 * hi) * 16 + (lane & 15)] = acc[i];
    }
    __syncthreads();

    const int pc = lane >> 3;      // pooled col within tile: 0..3
    const int oc = lane & 7;       // output channel
    float v0 = pl[(2 * pc    ) * 16 + oc];
    float v1 = pl[(2 * pc + 1) * 16 + oc];
    float v2 = pl[(8 + 2 * pc    ) * 16 + oc];
    float v3 = pl[(8 + 2 * pc + 1) * 16 + oc];
    float mx = fmaxf(fmaxf(v0, v1), fmaxf(v2, v3));
    float outv = fmaxf(mx + conv_b[oc], 0.0f);   // relu/pool commute (monotone)

    int px = 4 * tx + pc;
    if (px < PW)
        pooled[(((size_t)b * PH + py) * PW + px) * OC + oc] = outv;
}

// ---------------------------------------------------------------------------
// Kernel 2: FC1 partials.  [16 x 95048] @ [95048 x 32].  Bandwidth-bound on
// w1 (12.2 MB) -> keep lossless f32 WMMA.  K split across 1024 waves,
// atomicAdd partials into acc[16*32].
// ---------------------------------------------------------------------------
__global__ __launch_bounds__(32) void stn_fc1(
    const float* __restrict__ pooled,  // [16, 95048]
    const float* __restrict__ w1,      // [95048, 32]
    float* __restrict__ acc_out)       // [16*32]
{
    const int KC  = FLAT / 4;          // 23762
    const int PER = 24;                // ceil(23762/1024)
    int start = blockIdx.x * PER;
    int end   = start + PER;
    if (end > KC) end = KC;
    if (start >= KC) return;           // uniform per wave -> EXEC stays all-1

    const int lane = threadIdx.x & 31;
    const int hi   = lane >> 4;
    const int bM   = lane & 15;        // batch row (M)
    const int nn   = lane & 15;        // N within tile

    v8f acc0 = {};
    v8f acc1 = {};
    for (int k4 = start; k4 < end; ++k4) {
        int k = 4 * k4;
        v2f a = *reinterpret_cast<const v2f*>(pooled + (size_t)bM * FLAT + k + 2 * hi);
        v2f b0, b1;
        b0[0] = w1[(size_t)(k + 2 * hi    ) * 32 + nn];
        b0[1] = w1[(size_t)(k + 2 * hi + 1) * 32 + nn];
        b1[0] = w1[(size_t)(k + 2 * hi    ) * 32 + 16 + nn];
        b1[1] = w1[(size_t)(k + 2 * hi + 1) * 32 + 16 + nn];
        acc0 = __builtin_amdgcn_wmma_f32_16x16x4_f32(
            false, a, false, b0, (short)0, acc0, false, false);
        acc1 = __builtin_amdgcn_wmma_f32_16x16x4_f32(
            false, a, false, b1, (short)0, acc1, false, false);
    }

    #pragma unroll
    for (int i = 0; i < 8; ++i) {
        int M = i + 8 * hi;
        atomicAdd(&acc_out[M * 32 + nn],      acc0[i]);
        atomicAdd(&acc_out[M * 32 + 16 + nn], acc1[i]);
    }
}

// ---------------------------------------------------------------------------
// Kernel 3: h = relu(acc + b1); theta = h @ w2   -> theta[16,6]
// ---------------------------------------------------------------------------
__global__ void stn_theta(const float* __restrict__ acc,   // [16,32]
                          const float* __restrict__ b1,    // [32]
                          const float* __restrict__ w2,    // [32,6]
                          float* __restrict__ theta)       // [16,6]
{
    int t = threadIdx.x;
    if (t < 96) {
        int b = t / 6, j = t % 6;
        float s = 0.0f;
        #pragma unroll
        for (int k = 0; k < 32; ++k) {
            float h = fmaxf(acc[b * 32 + k] + b1[k], 0.0f);
            s += h * w2[k * 6 + j];
        }
        theta[t] = s;
    }
}

// ---------------------------------------------------------------------------
// Kernel 4: affine grid + bilinear sample (exact reference corner logic).
// ---------------------------------------------------------------------------
__global__ __launch_bounds__(256) void stn_sample(
    const float* __restrict__ x,       // [16,224,224,16]
    const float* __restrict__ theta,   // [16,6]
    float* __restrict__ out)           // [16,224,224,16]
{
    int idx = blockIdx.x * 256 + threadIdx.x;     // 802816 total, exact grid
    int xo = idx % IW;
    int t2 = idx / IW;
    int y  = t2 % IH;
    int b  = t2 / IH;

    float th0 = theta[b * 6 + 0], th1 = theta[b * 6 + 1], th2 = theta[b * 6 + 2];
    float th3 = theta[b * 6 + 3], th4 = theta[b * 6 + 4], th5 = theta[b * 6 + 5];

    float xs = -1.0f + (2.0f / (IW - 1)) * (float)xo;   // linspace(-1,1,224)
    float ys = -1.0f + (2.0f / (IH - 1)) * (float)y;

    float gx = (th0 * xs + th1 * ys + th2 + 1.0f) * 0.5f * (float)IW;
    float gy = (th3 * xs + th4 * ys + th5 + 1.0f) * 0.5f * (float)IH;

    int ix0 = (int)floorf(gx);
    ix0 = ix0 < 0 ? 0 : (ix0 > IW - 1 ? IW - 1 : ix0);
    int ix1 = ix0 + 1; if (ix1 > IW - 1) ix1 = IW - 1;
    int iy0 = (int)floorf(gy);
    iy0 = iy0 < 0 ? 0 : (iy0 > IH - 1 ? IH - 1 : iy0);
    int iy1 = iy0 + 1; if (iy1 > IH - 1) iy1 = IH - 1;

    float wx1 = (float)ix1 - gx, wx0 = gx - (float)ix0;
    float wy1 = (float)iy1 - gy, wy0 = gy - (float)iy0;
    float wa = wx1 * wy1, wb = wx1 * wy0, wc = wx0 * wy1, wd = wx0 * wy0;

    const float* pa = x + (((size_t)b * IH + iy0) * IW + ix0) * IC;
    const float* pb = x + (((size_t)b * IH + iy1) * IW + ix0) * IC;
    const float* pc = x + (((size_t)b * IH + iy0) * IW + ix1) * IC;
    const float* pd = x + (((size_t)b * IH + iy1) * IW + ix1) * IC;
    float* po = out + (size_t)idx * IC;

    #pragma unroll
    for (int c = 0; c < IC; c += 4) {
        float4 A = *reinterpret_cast<const float4*>(pa + c);
        float4 B = *reinterpret_cast<const float4*>(pb + c);
        float4 C = *reinterpret_cast<const float4*>(pc + c);
        float4 D = *reinterpret_cast<const float4*>(pd + c);
        float4 r;
        r.x = wa * A.x + wb * B.x + wc * C.x + wd * D.x;
        r.y = wa * A.y + wb * B.y + wc * C.y + wd * D.y;
        r.z = wa * A.z + wb * B.z + wc * C.z + wd * D.z;
        r.w = wa * A.w + wb * B.w + wc * C.w + wd * D.w;
        *reinterpret_cast<float4*>(po + c) = r;
    }
}

// ---------------------------------------------------------------------------
extern "C" void kernel_launch(void* const* d_in, const int* in_sizes, int n_in,
                              void* d_out, int out_size, void* d_ws, size_t ws_size,
                              hipStream_t stream) {
    const float* x      = (const float*)d_in[0];
    const float* conv_w = (const float*)d_in[1];
    const float* conv_b = (const float*)d_in[2];
    const float* w1     = (const float*)d_in[3];
    const float* b1     = (const float*)d_in[4];
    const float* w2     = (const float*)d_in[5];
    float* out = (float*)d_out;

    // Workspace layout (32-bit words):
    //   [0, 512)             : FC1 accumulator (atomics)
    //   [512, 1024)          : theta (96 used)
    //   [1024, 1024+6400)    : formatted f16 conv-weight fragments
    //   [7424, 7424+FLAT*16) : pooled activations (~6.08 MB)
    float*    acc    = (float*)d_ws;
    float*    theta  = acc + 512;
    unsigned* wfrag  = (unsigned*)(acc + 1024);
    float*    pooled = acc + 1024 + WFRAG_DW;

    stn_zero     <<<1, 512, 0, stream>>>(acc);
    stn_wfmt     <<<NCHUNK, 256, 0, stream>>>(conv_w, wfrag);
    stn_conv_pool<<<(BATCH * PH * 28) / 8, 256, 0, stream>>>(x, wfrag, conv_b, pooled);
    stn_fc1      <<<1024, 32, 0, stream>>>(pooled, w1, acc);
    stn_theta    <<<1, 128, 0, stream>>>(acc, b1, w2, theta);
    stn_sample   <<<(BATCH * IH * IW) / 256, 256, 0, stream>>>(x, theta, out);
}
